// CLRUTransitionModel_30305289240666
// MI455X (gfx1250) — matrix operand order repaired and checked by reading, not compiled, not run on hardware
//
#include <hip/hip_runtime.h>
#include <hip/hip_bf16.h>

typedef __attribute__((ext_vector_type(16))) _Float16 v16h;
typedef __attribute__((ext_vector_type(8)))  _Float16 v8h;
typedef __attribute__((ext_vector_type(8)))  float    v8f;

#define LATENT 64
#define UDIM   16
#define NYD    50
#define BTOT   2048
#define TSTEPS 128
#define DINP   96      /* DIN=81 padded to 96 (3 k-chunks of 32) */
#define KE     1056    /* composite K for E = 1024 + 16 bias rows + 16 pad */
#define KF     4160    /* composite K for F = 4096 + 64 bias rows */
#define EPSF   1e-5f

/* fp16 workspace layout (offsets in halves) */
#define OFF_W1   0        /* [128][96]  W1*bn_scale, transposed, K-padded */
#define OFF_W2   12288    /* [128][128] */
#define OFF_W3   28672    /* [64][128]  */
#define OFF_WNU  36864    /* [64][64]   */
#define OFF_WB   40960    /* [64][1056] repacked: [n][(k,u)] + bB rows    */
#define OFF_WD   108544   /* [64][1056] repacked: [y][(k,u)] + bD rows    */
#define OFF_WC   176128   /* [64][4160] repacked: [y][(k,n)] + bC rows    */
#define TOTAL_H  442368
#define SHIFT_BYTE_OFF ((size_t)TOTAL_H * 2)   /* then shift1[128], shift2[128] fp32 */

/* ---------------- prep: fp32 -> fp16 repack, BN fold ---------------- */
__global__ void clru_prep(const float* W1, const float* b1, const float* g1, const float* be1,
                          const float* m1, const float* v1,
                          const float* W2, const float* b2, const float* g2, const float* be2,
                          const float* m2, const float* v2,
                          const float* W3, const float* Wnu,
                          const float* WB, const float* bB,
                          const float* WC, const float* bC,
                          const float* WD, const float* bD,
                          _Float16* wsh, float* shift1, float* shift2)
{
  int stride = gridDim.x * blockDim.x;
  for (int idx = blockIdx.x * blockDim.x + threadIdx.x; idx < TOTAL_H; idx += stride) {
    float f = 0.f;
    if (idx < OFF_W2) {                       // W1h [128][96]
      int n = idx / 96, k = idx % 96;
      if (k < 81) f = W1[k * 128 + n] * g1[n] * rsqrtf(v1[n] + EPSF);
    } else if (idx < OFF_W3) {                // W2h [128][128]
      int j = idx - OFF_W2; int n = j / 128, k = j % 128;
      f = W2[k * 128 + n] * g2[n] * rsqrtf(v2[n] + EPSF);
    } else if (idx < OFF_WNU) {               // W3h [64][128]
      int j = idx - OFF_W3; int n = j / 128, k = j % 128;
      f = W3[k * 64 + n];
    } else if (idx < OFF_WB) {                // Wnuh [64][64]
      int j = idx - OFF_WNU; int n = j / 64, k = j % 64;
      f = Wnu[k * 64 + n];
    } else if (idx < OFF_WD) {                // WBp [64][1056]
      int j = idx - OFF_WB; int n = j / KE, kk = j % KE;
      if (kk < 1024)      f = WB[(kk >> 4) * 1024 + n * 16 + (kk & 15)];
      else if (kk < 1040) f = bB[n * 16 + (kk - 1024)];
    } else if (idx < OFF_WC) {                // WDp [64][1056], y padded 50->64
      int j = idx - OFF_WD; int y = j / KE, kk = j % KE;
      if (y < NYD) {
        if (kk < 1024)      f = WD[(kk >> 4) * 800 + y * 16 + (kk & 15)];
        else if (kk < 1040) f = bD[y * 16 + (kk - 1024)];
      }
    } else {                                  // WCp [64][4160]
      int j = idx - OFF_WC; int y = j / KF, kk = j % KF;
      if (y < NYD) {
        if (kk < 4096) f = WC[(kk >> 6) * 3200 + y * 64 + (kk & 63)];
        else           f = bC[y * 64 + (kk - 4096)];
      }
    }
    wsh[idx] = (_Float16)f;
  }
  for (int i = blockIdx.x * blockDim.x + threadIdx.x; i < 128; i += stride) {
    float s1 = g1[i] * rsqrtf(v1[i] + EPSF);
    float s2 = g2[i] * rsqrtf(v2[i] + EPSF);
    shift1[i] = (b1[i] - m1[i]) * s1 + be1[i];
    shift2[i] = (b2[i] - m2[i]) * s2 + be2[i];
  }
}

/* ---------------- WMMA fragment helpers (wave32, 16x16x32 f16) ---------------- */
__device__ __forceinline__ v8f wmma_f16(v16h a, v16h b, v8f c) {
  return __builtin_amdgcn_wmma_f32_16x16x32_f16(false, a, false, b, (short)0, c, false, false);
}

/* A fragment from row-major LDS [16][stride] halves (matches 16-bit A 16x32 layout) */
__device__ __forceinline__ v16h load_a_lds(const _Float16* a, int stride, int kc, int lane) {
  int row = lane & 15, hi = lane >> 4;
  const _Float16* p = a + row * stride + kc * 32 + hi * 8;
  v8h lo = *(const v8h*)p;
  v8h hh = *(const v8h*)(p + 16);
  v16h r;
#pragma unroll
  for (int i = 0; i < 8; ++i) { r[i] = lo[i]; r[8 + i] = hh[i]; }
  return r;
}

/* B fragment from K-contiguous transposed weights wt[col][kpad] (matches f16 B 32x16 layout) */
__device__ __forceinline__ v16h load_b_w(const _Float16* wt, int kpad, int n0, int kc, int lane) {
  int col = n0 + (lane & 15);
  int kb  = kc * 32 + (lane >> 4) * 8;
  const _Float16* p = wt + (size_t)col * kpad + kb;
  v8h lo = *(const v8h*)p;
  v8h hh = *(const v8h*)(p + 16);
  v16h r;
#pragma unroll
  for (int i = 0; i < 8; ++i) { r[i] = lo[i]; r[8 + i] = hh[i]; }
  return r;
}

/* E[b,(k,u)] = hz[b,k]*udt[b,u]; rows 1024..1039 carry udt (bias rows); rest pad */
__device__ __forceinline__ v16h make_e(const _Float16* hz, const _Float16* udt, int kc, int lane) {
  int row = lane & 15;
  int base = kc * 32 + (lane >> 4) * 8;
  v16h r;
#pragma unroll
  for (int g = 0; g < 2; ++g) {
    int bb = base + g * 16;
    if (bb < 1024) {
      _Float16 h = hz[row * 64 + (bb >> 4)];
      v8h uv = *(const v8h*)(udt + row * 16 + (bb & 15));
#pragma unroll
      for (int i = 0; i < 8; ++i) r[8 * g + i] = h * uv[i];
    } else if (bb < 1040) {
      v8h uv = *(const v8h*)(udt + row * 16 + (bb - 1024));
#pragma unroll
      for (int i = 0; i < 8; ++i) r[8 * g + i] = uv[i];
    } else {
#pragma unroll
      for (int i = 0; i < 8; ++i) r[8 * g + i] = (_Float16)0.f;
    }
  }
  return r;
}

/* F[b,(k,n)] = hz[b,k]*z1[b,n]; rows 4096..4159 carry z1 (bias rows) */
__device__ __forceinline__ v16h make_f(const _Float16* hz, const _Float16* z1h, int kc, int lane) {
  int row = lane & 15;
  int base = kc * 32 + (lane >> 4) * 8;
  v16h r;
#pragma unroll
  for (int g = 0; g < 2; ++g) {
    int bb = base + g * 16;
    if (bb < 4096) {
      _Float16 h = hz[row * 64 + (bb >> 6)];
      v8h zv = *(const v8h*)(z1h + row * 64 + (bb & 63));
#pragma unroll
      for (int i = 0; i < 8; ++i) r[8 * g + i] = h * zv[i];
    } else {
      v8h zv = *(const v8h*)(z1h + row * 64 + (bb - 4096));
#pragma unroll
      for (int i = 0; i < 8; ++i) r[8 * g + i] = zv[i];
    }
  }
  return r;
}

template <int KC>
__device__ __forceinline__ v8f gemm_lds(const _Float16* a, int astride,
                                        const _Float16* wt, int kpad, int n0, int lane) {
  v8f acc = {};
#pragma unroll
  for (int kc = 0; kc < KC; ++kc)
    acc = wmma_f16(load_a_lds(a, astride, kc, lane), load_b_w(wt, kpad, n0, kc, lane), acc);
  return acc;
}

/* -------- main: one block (4 waves) owns one 16-row batch tile for all T steps.
   Waves split the N-tiles of each phase; LDS holds the per-tile activations. -------- */
__global__ __launch_bounds__(128) void clru_main(const float* __restrict__ zt,
                                                 const float* __restrict__ dtv,
                                                 const float* __restrict__ U,
                                                 const float* __restrict__ b3,
                                                 const float* __restrict__ bnu,
                                                 const _Float16* __restrict__ wsh,
                                                 const float* __restrict__ shift1,
                                                 const float* __restrict__ shift2,
                                                 float* __restrict__ Zout,
                                                 float* __restrict__ Yout)
{
  __shared__ _Float16 sx  [16 * DINP];
  __shared__ _Float16 sh1 [16 * 128];
  __shared__ _Float16 sh2 [16 * 128];
  __shared__ _Float16 shz [16 * 64];
  __shared__ _Float16 sz1h[16 * 64];
  __shared__ _Float16 sudt[16 * 16];
  __shared__ float    szf [16 * 64];
  __shared__ float    syac[16 * 64];

  const int w       = threadIdx.x >> 5;   /* wave 0..3 */
  const int lane    = threadIdx.x & 31;
  const int rowbase = blockIdx.x * 16;

  /* init recurrent state from zt */
  for (int i = threadIdx.x; i < 16 * 64; i += 128) szf[i] = zt[(size_t)rowbase * 64 + i];
  __syncthreads();

  for (int t = 0; t < TSTEPS; ++t) {
    /* ---- build x = [z | u_t | dt | 0pad] and udt = u_t*dt (fp16, LDS) ---- */
    for (int i = threadIdx.x; i < 16 * DINP; i += 128) {
      int r = i / DINP, c = i % DINP;
      float f;
      if (c < 64)       f = szf[r * 64 + c];
      else if (c < 80)  f = U[(size_t)t * BTOT * UDIM + (size_t)(rowbase + r) * UDIM + (c - 64)];
      else if (c == 80) f = dtv[rowbase + r];
      else              f = 0.f;
      sx[i] = (_Float16)f;
    }
    for (int i = threadIdx.x; i < 16 * 16; i += 128) {
      int r = i / 16, u = i % 16;
      sudt[i] = (_Float16)(U[(size_t)t * BTOT * UDIM + (size_t)(rowbase + r) * UDIM + u] *
                           dtv[rowbase + r]);
    }
    if (t + 1 < TSTEPS)   /* warm near cache for next step's u_t (256 floats per tile) */
      __builtin_prefetch(U + (size_t)(t + 1) * BTOT * UDIM + (size_t)rowbase * UDIM +
                             (threadIdx.x & 127) * 2, 0, 3);
    __syncthreads();

    /* ---- h1 = relu(x@W1f + shift1); wave w does n-tiles {w, w+4}, shared A ---- */
    {
      int n0 = w * 16, n1 = (w + 4) * 16;
      v8f a0 = {}, a1 = {};
#pragma unroll
      for (int kc = 0; kc < 3; ++kc) {
        v16h va = load_a_lds(sx, DINP, kc, lane);
        a0 = wmma_f16(va, load_b_w(wsh + OFF_W1, DINP, n0, kc, lane), a0);
        a1 = wmma_f16(va, load_b_w(wsh + OFF_W1, DINP, n1, kc, lane), a1);
      }
      int c0 = n0 + (lane & 15), c1 = n1 + (lane & 15), rb = (lane >> 4) * 8;
      float s0 = shift1[c0], s1v = shift1[c1];
#pragma unroll
      for (int r = 0; r < 8; ++r) {
        float v0 = a0[r] + s0;  v0 = v0 > 0.f ? v0 : 0.f;
        float v1 = a1[r] + s1v; v1 = v1 > 0.f ? v1 : 0.f;
        sh1[(rb + r) * 128 + c0] = (_Float16)v0;
        sh1[(rb + r) * 128 + c1] = (_Float16)v1;
      }
    }
    __syncthreads();

    /* ---- h2 = relu(h1@W2f + shift2); wave w does n-tiles {w, w+4} ---- */
    {
      int n0 = w * 16, n1 = (w + 4) * 16;
      v8f a0 = {}, a1 = {};
#pragma unroll
      for (int kc = 0; kc < 4; ++kc) {
        v16h va = load_a_lds(sh1, 128, kc, lane);
        a0 = wmma_f16(va, load_b_w(wsh + OFF_W2, 128, n0, kc, lane), a0);
        a1 = wmma_f16(va, load_b_w(wsh + OFF_W2, 128, n1, kc, lane), a1);
      }
      int c0 = n0 + (lane & 15), c1 = n1 + (lane & 15), rb = (lane >> 4) * 8;
      float s0 = shift2[c0], s1v = shift2[c1];
#pragma unroll
      for (int r = 0; r < 8; ++r) {
        float v0 = a0[r] + s0;  v0 = v0 > 0.f ? v0 : 0.f;
        float v1 = a1[r] + s1v; v1 = v1 > 0.f ? v1 : 0.f;
        sh2[(rb + r) * 128 + c0] = (_Float16)v0;
        sh2[(rb + r) * 128 + c1] = (_Float16)v1;
      }
    }
    __syncthreads();

    /* ---- hz = h2@W3 + b3; wave w does n-tile w (2 independent acc chains) ---- */
    {
      int n0 = w * 16;
      v8f e0 = {}, e1 = {};
      {
        v16h va0 = load_a_lds(sh2, 128, 0, lane);
        v16h va1 = load_a_lds(sh2, 128, 1, lane);
        e0 = wmma_f16(va0, load_b_w(wsh + OFF_W3, 128, n0, 0, lane), e0);
        e1 = wmma_f16(va1, load_b_w(wsh + OFF_W3, 128, n0, 1, lane), e1);
        v16h va2 = load_a_lds(sh2, 128, 2, lane);
        v16h va3 = load_a_lds(sh2, 128, 3, lane);
        e0 = wmma_f16(va2, load_b_w(wsh + OFF_W3, 128, n0, 2, lane), e0);
        e1 = wmma_f16(va3, load_b_w(wsh + OFF_W3, 128, n0, 3, lane), e1);
      }
      int col = n0 + (lane & 15), rb = (lane >> 4) * 8;
      float bb = b3[col];
#pragma unroll
      for (int r = 0; r < 8; ++r)
        shz[(rb + r) * 64 + col] = (_Float16)(e0[r] + e1[r] + bb);
    }
    __syncthreads();

    /* ---- fused: nu tile + E@WBp tile + E@WDp tile for wave w (shared make_e) ---- */
    {
      int n0 = w * 16;
      v8f an = gemm_lds<2>(shz, 64, wsh + OFF_WNU, 64, n0, lane);
      v8f ae = {}, ad = {};
      for (int kc = 0; kc < KE / 32; ++kc) {
        v16h va = make_e(shz, sudt, kc, lane);
        ae = wmma_f16(va, load_b_w(wsh + OFF_WB, KE, n0, kc, lane), ae);
        ad = wmma_f16(va, load_b_w(wsh + OFF_WD, KE, n0, kc, lane), ad);
      }
      int col = n0 + (lane & 15), rb = (lane >> 4) * 8;
      float bn = bnu[col];
#pragma unroll
      for (int r = 0; r < 8; ++r) {
        float nu  = fminf(an[r] + bn, 2.0f);
        float lam = __expf(-__expf(nu));
        int rr = rb + r;
        float z1 = lam * szf[rr * 64 + col] + ae[r];
        szf[rr * 64 + col]  = z1;
        sz1h[rr * 64 + col] = (_Float16)z1;
        syac[rr * 64 + col] = ad[r];
        Zout[(size_t)t * BTOT * 64 + (size_t)(rowbase + rr) * 64 + col] = z1;
      }
    }
    __syncthreads();

    /* ---- y = F@WCp + y_d; wave w does y-tile w.
       Two accumulators over even/odd k-chunks -> independent WMMA chains. ---- */
    {
      int n0 = w * 16;
      v8f c0 = {}, c1 = {};
      for (int kc = 0; kc < KF / 32; kc += 2) {
        v16h va0 = make_f(shz, sz1h, kc, lane);
        v16h va1 = make_f(shz, sz1h, kc + 1, lane);
        c0 = wmma_f16(va0, load_b_w(wsh + OFF_WC, KF, n0, kc, lane), c0);
        c1 = wmma_f16(va1, load_b_w(wsh + OFF_WC, KF, n0, kc + 1, lane), c1);
      }
      int col = n0 + (lane & 15), rb = (lane >> 4) * 8;
      if (col < NYD) {
#pragma unroll
        for (int r = 0; r < 8; ++r)
          Yout[(size_t)t * BTOT * NYD + (size_t)(rowbase + rb + r) * NYD + col] =
              c0[r] + c1[r] + syac[(rb + r) * 64 + col];
      }
    }
    __syncthreads();
  }
}

extern "C" void kernel_launch(void* const* d_in, const int* in_sizes, int n_in,
                              void* d_out, int out_size, void* d_ws, size_t ws_size,
                              hipStream_t stream) {
  const float* zt  = (const float*)d_in[0];
  const float* dtv = (const float*)d_in[1];
  const float* U   = (const float*)d_in[2];
  const float* W1  = (const float*)d_in[3];
  const float* b1  = (const float*)d_in[4];
  const float* g1  = (const float*)d_in[5];
  const float* be1 = (const float*)d_in[6];
  const float* m1  = (const float*)d_in[7];
  const float* v1  = (const float*)d_in[8];
  const float* W2  = (const float*)d_in[9];
  const float* b2  = (const float*)d_in[10];
  const float* g2  = (const float*)d_in[11];
  const float* be2 = (const float*)d_in[12];
  const float* m2  = (const float*)d_in[13];
  const float* v2  = (const float*)d_in[14];
  const float* W3  = (const float*)d_in[15];
  const float* b3  = (const float*)d_in[16];
  const float* Wnu = (const float*)d_in[17];
  const float* bnu = (const float*)d_in[18];
  const float* WB  = (const float*)d_in[19];
  const float* bB  = (const float*)d_in[20];
  const float* WC  = (const float*)d_in[21];
  const float* bC  = (const float*)d_in[22];
  const float* WD  = (const float*)d_in[23];
  const float* bD  = (const float*)d_in[24];

  _Float16* wsh  = (_Float16*)d_ws;
  float* shift1  = (float*)((char*)d_ws + SHIFT_BYTE_OFF);
  float* shift2  = shift1 + 128;

  float* Zout = (float*)d_out;
  float* Yout = Zout + (size_t)TSTEPS * BTOT * 64;

  clru_prep<<<256, 256, 0, stream>>>(W1, b1, g1, be1, m1, v1,
                                     W2, b2, g2, be2, m2, v2,
                                     W3, Wnu, WB, bB, WC, bC, WD, bD,
                                     wsh, shift1, shift2);

  clru_main<<<BTOT / 16, 128, 0, stream>>>(zt, dtv, U, b3, bnu, wsh, shift1, shift2, Zout, Yout);
}
